// RPN_35708358099151
// MI455X (gfx1250) — matrix operand order repaired and checked by reading, not compiled, not run on hardware
//
#include <hip/hip_runtime.h>

typedef __attribute__((ext_vector_type(16))) __bf16 v16bf;
typedef __attribute__((ext_vector_type(8)))  float  v8f;
typedef unsigned int u32;
typedef u32 u32x4 __attribute__((ext_vector_type(4)));

struct B32x2 { u32x4 lo, hi; };

__device__ __forceinline__ unsigned short f2bf(float f) {
    u32 u = __builtin_bit_cast(u32, f);
    u += 0x7FFFu + ((u >> 16) & 1u);            // round-to-nearest-even
    return (unsigned short)(u >> 16);
}

// CDNA5 async copy: global -> LDS, 16B per lane, tracked by ASYNCcnt (no data VGPRs)
__device__ __forceinline__ void async_b128(unsigned ldsOff, const void* gaddr) {
    asm volatile("global_load_async_to_lds_b128 %0, %1, off"
                 :: "v"(ldsOff), "v"(gaddr) : "memory");
}
__device__ __forceinline__ void wait_async0() {
    asm volatile("s_wait_asynccnt 0x0" ::: "memory");
}

static constexpr int HH   = 64;
static constexpr int WW   = 64;
static constexpr int CIN  = 512;
static constexpr int COUT = 512;
static constexpr int KTOT = 9 * CIN;             // 4608
static constexpr int KT   = 32;                  // K per WMMA step (bf16)
static constexpr int NSTEP = KTOT / KT;          // 144
static constexpr int LDK  = KT + 8;              // LDS row pad: 80B stride, 16B aligned
static constexpr int NANCH = 36864;              // 64*64*9

// canonical Faster-RCNN anchors (base 16, ratios .5/1/2, scales 8/16/32)
__constant__ float ANC[9][4] = {
    { -84.f,  -40.f,  99.f,  55.f}, {-176.f,  -88.f, 191.f, 103.f}, {-360.f, -184.f, 375.f, 199.f},
    { -56.f,  -56.f,  71.f,  71.f}, {-120.f, -120.f, 135.f, 135.f}, {-248.f, -248.f, 263.f, 263.f},
    { -36.f,  -80.f,  51.f,  95.f}, { -80.f, -168.f,  95.f, 183.f}, {-168.f, -344.f, 183.f, 359.f}
};

// ---------------- prep kernels ----------------
__global__ void k_prep_act(const float* __restrict__ in, unsigned short* __restrict__ out) {
    int tid = blockIdx.x * 256 + threadIdx.x;              // over 4*64*64*512
    int c = tid & 511, pos = tid >> 9;
    int x = pos & 63, y = (pos >> 6) & 63, b = pos >> 12;
    out[tid] = f2bf(in[(((size_t)b * CIN + c) * HH + y) * WW + x]);
}

__global__ void k_prep_w2(const float* __restrict__ w, unsigned short* __restrict__ out) {
    int tid = blockIdx.x * 256 + threadIdx.x;              // over 512*4608
    int o = tid / KTOT, r = tid % KTOT;
    int tap = r >> 9, c = r & 511;
    out[tid] = f2bf(w[((size_t)o * CIN + c) * 9 + tap]);
}

__global__ void k_prep_whead(const float* __restrict__ wc, const float* __restrict__ wb,
                             const float* __restrict__ bc, const float* __restrict__ bb,
                             unsigned short* __restrict__ out, float* __restrict__ bias64,
                             unsigned short* __restrict__ zbuf) {
    int tid = blockIdx.x * 256 + threadIdx.x;              // over 64*512
    int n = tid >> 9, k = tid & 511;
    float v = 0.f;
    if (n < 18)      v = wc[n * 512 + k];
    else if (n < 54) v = wb[(n - 18) * 512 + k];
    out[tid] = f2bf(v);
    if (tid < 64) {
        float bv = 0.f;
        if (tid < 18)      bv = bc[tid];
        else if (tid < 54) bv = bb[tid - 18];
        bias64[tid] = bv;
        zbuf[tid] = 0;                                     // 128B zero pad source
    }
}

// ------- 3x3 conv: implicit GEMM, bf16 WMMA, async-to-LDS double buffering -------
// block tile 128(M) x 256(N), 8 waves, wave tile 64x64 -> acc[4][4], 16 WMMA / K-step
__global__ __launch_bounds__(256) void k_conv3x3(const unsigned short* __restrict__ act,
                                                 const unsigned short* __restrict__ w2,
                                                 const float* __restrict__ bias,
                                                 const unsigned short* __restrict__ zbuf,
                                                 unsigned short* __restrict__ out) {
    __shared__ unsigned short bufA[2][128][LDK];   // 20 KB
    __shared__ unsigned short bufB[2][256][LDK];   // 40 KB
    const int b = blockIdx.z;
    const int mBase = blockIdx.x * 128;
    const int nBase = blockIdx.y * 256;
    const int t = threadIdx.x;
    const int lane = t & 31, wid = t >> 5;
    const int wm = wid & 1, wn = wid >> 1;         // 2x4 wave grid, each 64x64
    const int lr = lane & 15, lh = lane >> 4;

    const v8f vz = {0.f, 0.f, 0.f, 0.f, 0.f, 0.f, 0.f, 0.f};
    v8f acc[4][4];
    for (int i = 0; i < 4; ++i) for (int j = 0; j < 4; ++j) acc[i][j] = vz;

    // staging map: thread t stages A row t>>1 (32B half-seg t&1) and full B row t (64B)
    const int ar = t >> 1;
    const int ah = t & 1;
    const int m = mBase + ar;
    const int y = m >> 6, x = m & 63;

    auto stage = [&](int buf, int kk) {
        const int tap = kk >> 9, c0 = kk & 511;
        const int ky = tap / 3 - 1, kx = tap % 3 - 1;
        // A tile: 128 x 32 bf16, zero-padded borders via zbuf redirect
        const int yy = y + ky, xx = x + kx;
        const bool ok = ((unsigned)yy < (unsigned)HH) & ((unsigned)xx < (unsigned)WW);
        const unsigned short* ga = ok
            ? act + (((size_t)(b * HH + yy) * WW + xx) * CIN + c0 + ah * 16)
            : zbuf;
        unsigned la = (unsigned)(uintptr_t)&bufA[buf][ar][ah * 16];
        async_b128(la,      ga);
        async_b128(la + 16, ga + 8);
        // B tile: 256 x 32 bf16 (weights, N-major K-contiguous)
        const unsigned short* gb = w2 + ((size_t)(nBase + t) * KTOT + kk);
        unsigned lb = (unsigned)(uintptr_t)&bufB[buf][t][0];
        async_b128(lb,      gb);
        async_b128(lb + 16, gb + 8);
        async_b128(lb + 32, gb + 16);
        async_b128(lb + 48, gb + 24);
    };

    stage(0, 0);
    wait_async0();
    __syncthreads();

    for (int s = 0; s < NSTEP; ++s) {
        const int cur = s & 1;
        if (s + 1 < NSTEP) stage(cur ^ 1, (s + 1) * KT);   // prefetch next tile async

        v16bf bfr[4];
#pragma unroll
        for (int j = 0; j < 4; ++j) {                      // B frag: lane N=lr, K=h*16..+15
            const int nl = wn * 64 + j * 16 + lr;
            B32x2 tmp;
            tmp.lo = *(const u32x4*)&bufB[cur][nl][lh * 16];
            tmp.hi = *(const u32x4*)&bufB[cur][nl][lh * 16 + 8];
            bfr[j] = __builtin_bit_cast(v16bf, tmp);
        }
#pragma unroll
        for (int i = 0; i < 4; ++i) {                      // A frag: lane M=lr, K=h*16..+15
            const int ml = wm * 64 + i * 16 + lr;
            B32x2 tmp;
            tmp.lo = *(const u32x4*)&bufA[cur][ml][lh * 16];
            tmp.hi = *(const u32x4*)&bufA[cur][ml][lh * 16 + 8];
            v16bf af = __builtin_bit_cast(v16bf, tmp);
#pragma unroll
            for (int j = 0; j < 4; ++j)
                acc[i][j] = __builtin_amdgcn_wmma_f32_16x16x32_bf16(
                    false, af, false, bfr[j], (short)0, acc[i][j], false, false);
        }
        wait_async0();
        __syncthreads();
    }

    // epilogue: bias + ReLU, NHWC bf16
    for (int j = 0; j < 4; ++j) {
        const int n = nBase + wn * 64 + j * 16 + lr;
        const float bn = bias[n];
        for (int i = 0; i < 4; ++i)
            for (int rr = 0; rr < 8; ++rr) {
                const int ml = wm * 64 + i * 16 + lh * 8 + rr;
                float v = acc[i][j][rr] + bn;
                v = v > 0.f ? v : 0.f;
                out[((size_t)b * 4096 + mBase + ml) * COUT + n] = f2bf(v);
            }
    }
}

// ---------------- 1x1 heads (cls 18 + bbox 36 packed to N=64) ----------------
__global__ __launch_bounds__(256) void k_heads(const unsigned short* __restrict__ actc,
                                               const unsigned short* __restrict__ wh,
                                               const float* __restrict__ bias64,
                                               float* __restrict__ headout) {
    __shared__ unsigned short lA[128][LDK];
    __shared__ unsigned short lB[64][LDK];
    const int mBase = blockIdx.x * 128;
    const int t = threadIdx.x;
    const int lane = t & 31, wid = t >> 5;
    const int lr = lane & 15, lh = lane >> 4;

    const v8f vz = {0.f, 0.f, 0.f, 0.f, 0.f, 0.f, 0.f, 0.f};
    v8f acc[4];
    for (int j = 0; j < 4; ++j) acc[j] = vz;

    const int r = t >> 1, hh = t & 1;
    for (int kk = 0; kk < 512; kk += KT) {
        {
            const u32x4* p = (const u32x4*)(actc + ((size_t)(mBase + r) * 512 + kk + hh * 16));
            u32x4* d = (u32x4*)&lA[r][hh * 16];
            d[0] = p[0]; d[1] = p[1];
        }
        if (t < 128) {
            const int rb = t >> 1, hb = t & 1;
            const u32x4* p = (const u32x4*)(wh + ((size_t)rb * 512 + kk + hb * 16));
            u32x4* d = (u32x4*)&lB[rb][hb * 16];
            d[0] = p[0]; d[1] = p[1];
        }
        __syncthreads();

        v16bf a;
        {
            const int ml = wid * 16 + lr;
            B32x2 tmp;
            tmp.lo = *(const u32x4*)&lA[ml][lh * 16];
            tmp.hi = *(const u32x4*)&lA[ml][lh * 16 + 8];
            a = __builtin_bit_cast(v16bf, tmp);
        }
#pragma unroll
        for (int j = 0; j < 4; ++j) {
            const int nl = j * 16 + lr;
            B32x2 tmp;
            tmp.lo = *(const u32x4*)&lB[nl][lh * 16];
            tmp.hi = *(const u32x4*)&lB[nl][lh * 16 + 8];
            v16bf bb = __builtin_bit_cast(v16bf, tmp);
            acc[j] = __builtin_amdgcn_wmma_f32_16x16x32_bf16(
                false, a, false, bb, (short)0, acc[j], false, false);
        }
        __syncthreads();
    }
    for (int j = 0; j < 4; ++j) {
        const int n = j * 16 + lr;
        const float bn = bias64[n];
        for (int rr = 0; rr < 8; ++rr) {
            const int ml = wid * 16 + lh * 8 + rr;
            headout[(size_t)(mBase + ml) * 64 + n] = acc[j][rr] + bn;
        }
    }
}

// ---------------- softmax + anchor decode + clip ----------------
__global__ void k_proposals(const float* __restrict__ headout, const float* __restrict__ im_info,
                            float* __restrict__ scores, float* __restrict__ boxes) {
    int tid = blockIdx.x * 256 + threadIdx.x;
    if (tid >= 4 * NANCH) return;
    int b = tid / NANCH, remi = tid % NANCH;
    int pos = remi / 9, a = remi % 9;
    int x = pos & 63, y = pos >> 6;
    const float* h = headout + (size_t)(b * 4096 + pos) * 64;
    float s0 = h[a], s1 = h[9 + a];
    float fg = 1.f / (1.f + __expf(s0 - s1));
    float d0 = h[18 + a * 4 + 0], d1 = h[18 + a * 4 + 1];
    float d2 = h[18 + a * 4 + 2], d3 = h[18 + a * 4 + 3];
    float ax1 = ANC[a][0] + x * 16.f, ay1 = ANC[a][1] + y * 16.f;
    float ax2 = ANC[a][2] + x * 16.f, ay2 = ANC[a][3] + y * 16.f;
    float aw = ax2 - ax1 + 1.f, ah = ay2 - ay1 + 1.f;
    float acx = ax1 + 0.5f * aw, acy = ay1 + 0.5f * ah;
    float pcx = d0 * aw + acx, pcy = d1 * ah + acy;
    float pw = __expf(d2) * aw, ph = __expf(d3) * ah;
    float imh = im_info[b * 3 + 0], imw = im_info[b * 3 + 1];
    float x1 = fminf(fmaxf(pcx - 0.5f * pw, 0.f), imw - 1.f);
    float y1 = fminf(fmaxf(pcy - 0.5f * ph, 0.f), imh - 1.f);
    float x2 = fminf(fmaxf(pcx + 0.5f * pw, 0.f), imw - 1.f);
    float y2 = fminf(fmaxf(pcy + 0.5f * ph, 0.f), imh - 1.f);
    scores[tid] = fg;
    ((float4*)boxes)[tid] = make_float4(x1, y1, x2, y2);
}

// ------- greedy NMS: scores in 147KB dynamic LDS (320KB/WGP on CDNA5) -------
__global__ __launch_bounds__(1024) void k_nms(const float* __restrict__ scores,
                                              const float* __restrict__ boxes,
                                              float* __restrict__ out) {
    extern __shared__ char smemraw[];
    float* sc = (float*)smemraw;            // 36864
    float* rv = sc + NANCH;                 // 1024
    int*   ri = (int*)(rv + 1024);          // 1024
    float* wb = (float*)(ri + 1024);        // 4 box + valid flag
    const int b = blockIdx.x, t = threadIdx.x;
    const float NEG = -1e30f;
    for (int i = t; i < NANCH; i += 1024) sc[i] = scores[b * NANCH + i];
    __syncthreads();
    const float4* bx = (const float4*)boxes + (size_t)b * NANCH;
    for (int it = 0; it < 300; ++it) {
        float v = NEG; int id = 0;
        for (int i = t; i < NANCH; i += 1024) { float s = sc[i]; if (s > v) { v = s; id = i; } }
        rv[t] = v; ri[t] = id;
        __syncthreads();
        for (int s = 512; s > 0; s >>= 1) {
            if (t < s && rv[t + s] > rv[t]) { rv[t] = rv[t + s]; ri[t] = ri[t + s]; }
            __syncthreads();
        }
        if (t == 0) {
            int best = ri[0];
            float bv = rv[0];
            float* o = out + ((size_t)b * 300 + it) * 5;
            if (bv > 0.5f * NEG) {
                float4 bb = bx[best];
                wb[0] = bb.x; wb[1] = bb.y; wb[2] = bb.z; wb[3] = bb.w; wb[4] = 1.f;
                sc[best] = NEG;
                o[0] = (float)b; o[1] = bb.x; o[2] = bb.y; o[3] = bb.z; o[4] = bb.w;
            } else {
                wb[4] = 0.f;
                o[0] = (float)b; o[1] = 0.f; o[2] = 0.f; o[3] = 0.f; o[4] = 0.f;
            }
        }
        __syncthreads();
        if (wb[4] > 0.5f) {
            float bx1 = wb[0], by1 = wb[1], bx2 = wb[2], by2 = wb[3];
            float barea = (bx2 - bx1 + 1.f) * (by2 - by1 + 1.f);
            for (int i = t; i < NANCH; i += 1024) {
                if (sc[i] > 0.5f * NEG) {
                    float4 c = bx[i];
                    float xx1 = fmaxf(bx1, c.x), yy1 = fmaxf(by1, c.y);
                    float xx2 = fminf(bx2, c.z), yy2 = fminf(by2, c.w);
                    float iw = fmaxf(xx2 - xx1 + 1.f, 0.f), ih = fmaxf(yy2 - yy1 + 1.f, 0.f);
                    float inter = iw * ih;
                    float carea = (c.z - c.x + 1.f) * (c.w - c.y + 1.f);
                    if (inter > 0.7f * (barea + carea - inter)) sc[i] = NEG;
                }
            }
        }
        __syncthreads();
    }
}

extern "C" void kernel_launch(void* const* d_in, const int* in_sizes, int n_in,
                              void* d_out, int out_size, void* d_ws, size_t ws_size,
                              hipStream_t stream) {
    const float* base_feat = (const float*)d_in[0];
    const float* im_info   = (const float*)d_in[1];
    const float* W_conv    = (const float*)d_in[4];
    const float* b_conv    = (const float*)d_in[5];
    const float* W_cls     = (const float*)d_in[6];
    const float* b_cls     = (const float*)d_in[7];
    const float* W_bbox    = (const float*)d_in[8];
    const float* b_bbox    = (const float*)d_in[9];

    char* ws = (char*)d_ws;
    unsigned short* act    = (unsigned short*)(ws + 0);          // 16 MB  NHWC bf16
    unsigned short* w2     = (unsigned short*)(ws + 16777216);   // 4.5 MB [O][9*512] bf16
    unsigned short* cvo    = (unsigned short*)(ws + 21495808);   // 16 MB  conv out NHWC bf16
    unsigned short* wh     = (unsigned short*)(ws + 38273024);   // 64 KB  [64][512] bf16
    float* bias64          = (float*)(ws + 38338560);            // 256 B
    unsigned short* zbuf   = (unsigned short*)(ws + 38338816);   // 128 B zero pad
    float* headout         = (float*)(ws + 38339072);            // 4 MB   [16384][64] f32
    float* scoresB         = (float*)(ws + 42533376);            // 576 KB
    float* boxesB          = (float*)(ws + 43123200);            // 2.25 MB float4

    k_prep_act  <<<32768, 256, 0, stream>>>(base_feat, act);
    k_prep_w2   <<<9216,  256, 0, stream>>>(W_conv, w2);
    k_prep_whead<<<128,   256, 0, stream>>>(W_cls, W_bbox, b_cls, b_bbox, wh, bias64, zbuf);
    k_conv3x3   <<<dim3(32, 2, 4), 256, 0, stream>>>(act, w2, b_conv, zbuf, cvo);
    k_heads     <<<128,   256, 0, stream>>>(cvo, wh, bias64, headout);
    k_proposals <<<576,   256, 0, stream>>>(headout, im_info, scoresB, boxesB);
    size_t smem = (size_t)NANCH * 4 + 1024 * 4 + 1024 * 4 + 32;
    k_nms       <<<4, 1024, smem, stream>>>(scoresB, boxesB, (float*)d_out);
}
